// SparseAutoencoder_11287174053889
// MI455X (gfx1250) — matrix operand order repaired and checked by reading, not compiled, not run on hardware
//
#include <hip/hip_runtime.h>
#include <hip/hip_bf16.h>
#include <math.h>

// Problem sizes (fixed by the reference).
constexpr int kB = 8192;    // batch rows
constexpr int kD = 1536;    // input dim
constexpr int kH = 16384;   // latent dim
constexpr int MAXNZ = 128;  // per-row compacted nonzero cap (k=64 + rare ties)

typedef __attribute__((ext_vector_type(16))) __bf16 v16bf;
typedef __attribute__((ext_vector_type(8)))  __bf16 v8bf;
typedef __attribute__((ext_vector_type(4)))  __bf16 v4bf;
typedef __attribute__((ext_vector_type(8)))  float  v8f;

// ---- CDNA5 async global->LDS staging (no VGPR payload, tracked by ASYNCcnt)
#if defined(__has_builtin)
#if __has_builtin(__builtin_amdgcn_global_load_async_to_lds_b128)
#define USE_ASYNC_STAGE 1
#endif
#endif
#ifndef USE_ASYNC_STAGE
#define USE_ASYNC_STAGE 0
#endif

typedef int v4i __attribute__((vector_size(16)));          // b128 payload
typedef v4i __attribute__((address_space(1))) gv4i;        // global (AS1)
typedef v4i __attribute__((address_space(3))) lv4i;        // LDS (AS3)

__device__ __forceinline__ void stage_async_b128(const void* g, void* l) {
#if USE_ASYNC_STAGE
  __builtin_amdgcn_global_load_async_to_lds_b128(
      (gv4i*)g, (lv4i*)l, /*offset=*/0, /*cpol=*/0);
#else
  (void)g; (void)l;
#endif
}

__device__ __forceinline__ void wait_async0() {
#if defined(__has_builtin) && __has_builtin(__builtin_amdgcn_s_wait_asynccnt)
  __builtin_amdgcn_s_wait_asynccnt(0);
#else
  asm volatile("s_wait_asynccnt 0x0" ::: "memory");
#endif
}

// ---------------------------------------------------------------- init stats
__global__ void init_accum_kernel(double* acc_abs, double* acc_sq,
                                  unsigned long long* acc_nnz) {
  if (threadIdx.x == 0) { *acc_abs = 0.0; *acc_sq = 0.0; *acc_nnz = 0ull; }
}

// ----------------------------------------------- row normalize -> bf16 xn-b
__global__ __launch_bounds__(256) void rownorm_kernel(
    const float* __restrict__ x, const float* __restrict__ pre_bias,
    __bf16* __restrict__ xnb, float* __restrict__ mu, float* __restrict__ sd) {
  const int row = blockIdx.x;
  const int tid = threadIdx.x;
  const float* xr = x + (size_t)row * kD;
  float v[6];
  float s = 0.f, q = 0.f;
#pragma unroll
  for (int j = 0; j < 6; ++j) {
    v[j] = xr[tid + j * 256];
    s += v[j];
    q += v[j] * v[j];
  }
  __shared__ float rs[256], rq[256];
  rs[tid] = s; rq[tid] = q;
  __syncthreads();
  for (int off = 128; off > 0; off >>= 1) {
    if (tid < off) { rs[tid] += rs[tid + off]; rq[tid] += rq[tid + off]; }
    __syncthreads();
  }
  __shared__ float s_mu, s_scale;
  if (tid == 0) {
    float m   = rs[0] / (float)kD;
    float var = (rq[0] - rs[0] * rs[0] / (float)kD) / (float)(kD - 1);
    float st  = sqrtf(fmaxf(var, 0.f));
    s_mu = m; s_scale = 1.f / (st + 1e-5f);
    mu[row] = m; sd[row] = st;
  }
  __syncthreads();
  const float m = s_mu, sc = s_scale;
  __bf16* orow = xnb + (size_t)row * kD;
#pragma unroll
  for (int j = 0; j < 6; ++j) {
    int i = tid + j * 256;
    orow[i] = (__bf16)((v[j] - m) * sc - pre_bias[i]);
  }
}

// ------------------------------------------------------------- W_enc -> bf16
__global__ void wenc_bf16_kernel(const float* __restrict__ w,
                                 __bf16* __restrict__ wb, int n4) {
  int i = blockIdx.x * blockDim.x + threadIdx.x;
  if (i < n4) {
    float4 f = ((const float4*)w)[i];
    v4bf o;
    o[0] = (__bf16)f.x; o[1] = (__bf16)f.y;
    o[2] = (__bf16)f.z; o[3] = (__bf16)f.w;
    ((v4bf*)wb)[i] = o;
  }
}

// ------------------------------------------- W_dec (D,H) -> W_decT (H,D) f32
__global__ __launch_bounds__(256) void transpose_kernel(
    const float* __restrict__ src, float* __restrict__ dst) {
  __shared__ float tile[32][33];
  int hx = blockIdx.x * 32 + threadIdx.x;  // H index
  int d0 = blockIdx.y * 32;                // D tile base
#pragma unroll
  for (int j = 0; j < 32; j += 8)
    tile[threadIdx.y + j][threadIdx.x] =
        src[(size_t)(d0 + threadIdx.y + j) * kH + hx];
  __syncthreads();
  int dx = d0 + threadIdx.x;
  int h0 = blockIdx.x * 32;
#pragma unroll
  for (int j = 0; j < 32; j += 8)
    dst[(size_t)(h0 + threadIdx.y + j) * kD + dx] =
        tile[threadIdx.x][threadIdx.y + j];
}

// --------------------------- encoder GEMM: h = relu(xnb @ W_encT + lat_bias)
// Block tile 64(M) x 256(N), 8 waves in 2x4; each wave owns 32x64 via 2x4
// WMMA subtiles (8 v_wmma per K-step). Double-buffered LDS; the next K-step
// tile is staged with GLOBAL_LOAD_ASYNC_TO_LDS_B128 (DMA concurrent with the
// WMMAs, zero VGPR payload -> no spill), synced by s_wait_asynccnt + barrier.
__global__ __launch_bounds__(256, 1) void enc_gemm_kernel(
    const __bf16* __restrict__ A,   // [kB][kD] bf16
    const __bf16* __restrict__ Bw,  // [kH][kD] bf16
    const float* __restrict__ latent_bias,
    float* __restrict__ Hout) {     // [kB][kH] f32
  constexpr int LDK = 48;  // padded K stride (elems): 96B rows, 16B aligned
  constexpr int BM = 64, BN = 256, BK = 32;
  __shared__ __bf16 As[2][BM * LDK];   // 2 x  6 KB
  __shared__ __bf16 Bs[2][BN * LDK];   // 2 x 24 KB
  const int tid  = threadIdx.x;
  const int lane = tid & 31;
  const int wave = tid >> 5;
  const int wm = wave >> 2;  // 0..1 -> M offset wm*32
  const int wn = wave & 3;   // 0..3 -> N offset wn*64
  const int mBase = blockIdx.y * BM;
  const int nBase = blockIdx.x * BN;

  v8f acc[2][4];
#pragma unroll
  for (int s = 0; s < 2; ++s)
#pragma unroll
    for (int t = 0; t < 4; ++t)
#pragma unroll
      for (int e = 0; e < 8; ++e) acc[s][t][e] = 0.f;

  const int mloc = lane & 15;
  const int hi   = lane >> 4;

  // staging assignment: 16B chunks; A = 1/thread, B = 4/thread
  const int srow = tid >> 2;          // 0..63
  const int sseg = tid & 3;           // 0..3  (seg*8 bf16)
  const __bf16* aSrc = A  + (size_t)(mBase + srow) * kD + sseg * 8;
  const __bf16* bSrc = Bw + (size_t)(nBase + srow) * kD + sseg * 8;
  __bf16* aDst0 = &As[0][srow * LDK + sseg * 8];
  __bf16* bDst0 = &Bs[0][srow * LDK + sseg * 8];
  const int aBufStride = BM * LDK;           // elems between As buffers
  const int bBufStride = BN * LDK;           // elems between Bs buffers

#if USE_ASYNC_STAGE
  // ---- preload K-step 0 into buffer 0 via async DMA
  stage_async_b128(aSrc, aDst0);
#pragma unroll
  for (int rep = 0; rep < 4; ++rep)
    stage_async_b128(bSrc + (size_t)rep * 64 * kD, bDst0 + rep * 64 * LDK);
  wait_async0();
  __syncthreads();
#else
  {  // fallback: register staging
    uint4 aReg = *(const uint4*)aSrc;
    uint4 bReg[4];
#pragma unroll
    for (int rep = 0; rep < 4; ++rep)
      bReg[rep] = *(const uint4*)(bSrc + (size_t)rep * 64 * kD);
    *(uint4*)aDst0 = aReg;
#pragma unroll
    for (int rep = 0; rep < 4; ++rep)
      *(uint4*)(bDst0 + rep * 64 * LDK) = bReg[rep];
    __syncthreads();
  }
#endif

  int cur = 0;
  for (int kk = 0; kk < kD; kk += BK) {
    const int nxt = cur ^ 1;
    const bool more = (kk + BK) < kD;

#if USE_ASYNC_STAGE
    if (more) {  // DMA next tile; runs underneath the WMMAs below
      stage_async_b128(aSrc + kk + BK, aDst0 + nxt * aBufStride);
#pragma unroll
      for (int rep = 0; rep < 4; ++rep)
        stage_async_b128(bSrc + (size_t)rep * 64 * kD + kk + BK,
                         bDst0 + nxt * bBufStride + rep * 64 * LDK);
    }
#else
    if (more) {  // fallback: load + store to LDS before WMMAs (short liveness)
      uint4 aReg = *(const uint4*)(aSrc + kk + BK);
      uint4 bReg[4];
#pragma unroll
      for (int rep = 0; rep < 4; ++rep)
        bReg[rep] = *(const uint4*)(bSrc + (size_t)rep * 64 * kD + kk + BK);
      *(uint4*)(aDst0 + nxt * aBufStride) = aReg;
#pragma unroll
      for (int rep = 0; rep < 4; ++rep)
        *(uint4*)(bDst0 + nxt * bBufStride + rep * 64 * LDK) = bReg[rep];
    }
#endif

    // fragments from current buffer
    v16bf af[2], bfr[4];
#pragma unroll
    for (int s = 0; s < 2; ++s) {  // A frag: lane m=lane&15, K chunks of 8
      int ar = wm * 32 + s * 16 + mloc;
      union { v8bf h[2]; v16bf v; } u;
      u.h[0] = *(const v8bf*)&As[cur][ar * LDK + hi * 8];
      u.h[1] = *(const v8bf*)&As[cur][ar * LDK + hi * 8 + 16];
      af[s] = u.v;
    }
#pragma unroll
    for (int t = 0; t < 4; ++t) {  // B frag: lane n=lane&15, K chunks of 8
      int br = wn * 64 + t * 16 + mloc;
      union { v8bf h[2]; v16bf v; } u;
      u.h[0] = *(const v8bf*)&Bs[cur][br * LDK + hi * 8];
      u.h[1] = *(const v8bf*)&Bs[cur][br * LDK + hi * 8 + 16];
      bfr[t] = u.v;
    }
#pragma unroll
    for (int s = 0; s < 2; ++s)
#pragma unroll
      for (int t = 0; t < 4; ++t)
        acc[s][t] = __builtin_amdgcn_wmma_f32_16x16x32_bf16(
            false, af[s], false, bfr[t], (short)0, acc[s][t], false, false);

#if USE_ASYNC_STAGE
    wait_async0();  // my wave's DMA into 'nxt' done (counts are per-wave)
#endif
    __syncthreads();
    cur = nxt;
  }

  // C/D layout: VGPR r -> M = r (lanes 0-15) or r+8 (lanes 16-31), N = lane&15
  const int nloc = lane & 15;
  const int rofs = (lane < 16) ? 0 : 8;
#pragma unroll
  for (int s = 0; s < 2; ++s)
#pragma unroll
    for (int t = 0; t < 4; ++t) {
      int col = nBase + wn * 64 + t * 16 + nloc;
      float lb = latent_bias[col];
#pragma unroll
      for (int r = 0; r < 8; ++r) {
        int rowg = mBase + wm * 32 + s * 16 + r + rofs;
        float vv = acc[s][t][r] + lb;
        Hout[(size_t)rowg * kH + col] = vv > 0.f ? vv : 0.f;
      }
    }
}

// -------- per-row exact top-k via 4x8-bit radix select on float bit patterns
// (values are non-negative post-ReLU, so uint order == float order).
__global__ __launch_bounds__(256) void topk_kernel(
    float* __restrict__ Hbase, const int* __restrict__ kp,
    float* __restrict__ nzv, int* __restrict__ nzi, int* __restrict__ nzc,
    double* __restrict__ acc_abs, double* __restrict__ acc_sq,
    unsigned long long* __restrict__ acc_nnz) {
  const int row = blockIdx.x;
  const int tid = threadIdx.x;
  float* hr = Hbase + (size_t)row * kH;
  __shared__ unsigned hist[256];
  __shared__ unsigned s_prefix;
  __shared__ int s_k;
  if (tid == 0) { s_prefix = 0u; s_k = *kp; }
  __syncthreads();
  unsigned mask = 0u;
#pragma unroll
  for (int p = 0; p < 4; ++p) {
    const int shift = 24 - 8 * p;
    const unsigned prefix = s_prefix;
    hist[tid] = 0u;
    __syncthreads();
    for (int i = tid; i < kH; i += 256) {
      unsigned key = __float_as_uint(hr[i]);
      if ((key & mask) == prefix) atomicAdd(&hist[(key >> shift) & 255u], 1u);
    }
    __syncthreads();
    if (tid == 0) {
      int kk = s_k;
      unsigned cum = 0;
      int b = 255;
      for (; b >= 0; --b) {
        unsigned c = hist[b];
        if (cum + c >= (unsigned)kk) {
          s_prefix = prefix | ((unsigned)b << shift);
          s_k = kk - (int)cum;
          break;
        }
        cum += c;
      }
      if (b < 0) { s_prefix = prefix; }  // safety; k <= kH so unreachable
    }
    mask |= (0xFFu << shift);
    __syncthreads();
  }
  const unsigned tb = s_prefix;  // bits of exact kth-largest value
  __shared__ unsigned s_nz;
  if (tid == 0) s_nz = 0u;
  __syncthreads();
  double la = 0.0, lq = 0.0;
  unsigned lcnt = 0;
  for (int i = tid; i < kH; i += 256) {
    float v = hr[i];
    float o = (__float_as_uint(v) >= tb) ? v : 0.f;  // keeps ties, like ref
    hr[i] = o;
    if (o != 0.f) {
      unsigned pos = atomicAdd(&s_nz, 1u);
      if (pos < MAXNZ) {
        nzv[(size_t)row * MAXNZ + pos] = o;
        nzi[(size_t)row * MAXNZ + pos] = i;
      }
      la += (double)o;
      lq += (double)o * (double)o;
      ++lcnt;
    }
  }
  __shared__ double rd[256];
  rd[tid] = la; __syncthreads();
  for (int off = 128; off > 0; off >>= 1) {
    if (tid < off) rd[tid] += rd[tid + off];
    __syncthreads();
  }
  double tot_abs = rd[0]; __syncthreads();
  rd[tid] = lq; __syncthreads();
  for (int off = 128; off > 0; off >>= 1) {
    if (tid < off) rd[tid] += rd[tid + off];
    __syncthreads();
  }
  double tot_sq = rd[0]; __syncthreads();
  rd[tid] = (double)lcnt; __syncthreads();
  for (int off = 128; off > 0; off >>= 1) {
    if (tid < off) rd[tid] += rd[tid + off];
    __syncthreads();
  }
  if (tid == 0) {
    atomicAdd(acc_abs, tot_abs);
    atomicAdd(acc_sq, tot_sq);
    atomicAdd(acc_nnz, (unsigned long long)rd[0]);
    unsigned c = s_nz;
    nzc[row] = (int)(c < (unsigned)MAXNZ ? c : (unsigned)MAXNZ);
  }
}

// -------------- sparse decoder: recon = (h_sp @ W_decT + pre_bias)*std + mu
// W_decT (H,D) is 100 MB -> fully L2-resident; reads are coalesced over D.
__global__ __launch_bounds__(256) void dec_kernel(
    const float* __restrict__ WdT, const float* __restrict__ nzv,
    const int* __restrict__ nzi, const int* __restrict__ nzc,
    const float* __restrict__ pre_bias, const float* __restrict__ mu,
    const float* __restrict__ sd, float* __restrict__ recon) {
  const int row = blockIdx.x;
  const int tid = threadIdx.x;
  __shared__ float sv[MAXNZ];
  __shared__ int si[MAXNZ];
  const int cnt = nzc[row];
  if (tid < cnt) {
    sv[tid] = nzv[(size_t)row * MAXNZ + tid];
    si[tid] = nzi[(size_t)row * MAXNZ + tid];
  }
  __syncthreads();
  float acc[6];
#pragma unroll
  for (int j = 0; j < 6; ++j) acc[j] = 0.f;
  for (int i = 0; i < cnt; ++i) {
    float v = sv[i];
    const float* wr = WdT + (size_t)si[i] * kD;
#pragma unroll
    for (int j = 0; j < 6; ++j) acc[j] = fmaf(v, wr[tid + j * 256], acc[j]);
  }
  const float m = mu[row], st = sd[row];
  float* rr = recon + (size_t)row * kD;
#pragma unroll
  for (int j = 0; j < 6; ++j) {
    int d = tid + j * 256;
    rr[d] = (acc[j] + pre_bias[d]) * st + m;
  }
}

// ------------------------------------------------------------ final scalars
__global__ void finalize_kernel(const double* acc_abs, const double* acc_sq,
                                const unsigned long long* acc_nnz,
                                float* out3) {
  if (threadIdx.x != 0) return;
  const double N = (double)kB * (double)kH;
  const double nnz = (double)*acc_nnz;
  const double sa = *acc_abs, sq = *acc_sq;
  out3[0] = (float)((N - nnz) / N);          // sparsity ratio
  const double mean = sa / N;
  out3[1] = (float)mean;                     // mean |activation|
  double var = (sq - N * mean * mean) / (N - 1.0);
  out3[2] = (float)(var > 0.0 ? sqrt(var) : 0.0);
}

// ---------------------------------------------------------------------------
extern "C" void kernel_launch(void* const* d_in, const int* in_sizes, int n_in,
                              void* d_out, int out_size, void* d_ws,
                              size_t ws_size, hipStream_t stream) {
  (void)in_sizes; (void)n_in; (void)out_size; (void)ws_size;
  const float* x           = (const float*)d_in[0];
  const float* W_enc       = (const float*)d_in[1];
  const float* W_dec       = (const float*)d_in[2];
  const float* pre_bias    = (const float*)d_in[3];
  const float* latent_bias = (const float*)d_in[4];
  const int*   kp          = (const int*)d_in[5];

  // workspace carve-up (256B aligned)
  char* ws = (char*)d_ws;
  size_t off = 0;
  auto carve = [&](size_t bytes) {
    void* p = ws + off;
    off = (off + bytes + 255) & ~(size_t)255;
    return p;
  };
  __bf16* xnb   = (__bf16*)carve((size_t)kB * kD * sizeof(__bf16)); //  25 MB
  __bf16* wencb = (__bf16*)carve((size_t)kH * kD * sizeof(__bf16)); //  50 MB
  float*  wdT   = (float*)carve((size_t)kH * kD * sizeof(float));   // 100 MB
  float*  mu    = (float*)carve((size_t)kB * sizeof(float));
  float*  sd    = (float*)carve((size_t)kB * sizeof(float));
  float*  nzv   = (float*)carve((size_t)kB * MAXNZ * sizeof(float));
  int*    nzi   = (int*)carve((size_t)kB * MAXNZ * sizeof(int));
  int*    nzc   = (int*)carve((size_t)kB * sizeof(int));
  double* acc_abs = (double*)carve(sizeof(double));
  double* acc_sq  = (double*)carve(sizeof(double));
  unsigned long long* acc_nnz = (unsigned long long*)carve(sizeof(unsigned long long));

  float* recon = (float*)d_out;                                  // [kB][kD]
  float* hsp   = (float*)d_out + (size_t)kB * kD;                // [kB][kH]
  float* out3  = (float*)d_out + (size_t)kB * kD + (size_t)kB * kH;  // 3

  init_accum_kernel<<<1, 64, 0, stream>>>(acc_abs, acc_sq, acc_nnz);
  rownorm_kernel<<<kB, 256, 0, stream>>>(x, pre_bias, xnb, mu, sd);
  wenc_bf16_kernel<<<(kH * kD / 4 + 255) / 256, 256, 0, stream>>>(
      W_enc, wencb, kH * kD / 4);
  transpose_kernel<<<dim3(kH / 32, kD / 32), dim3(32, 8), 0, stream>>>(W_dec, wdT);
  enc_gemm_kernel<<<dim3(kH / 256, kB / 64), 256, 0, stream>>>(
      xnb, wencb, latent_bias, hsp);
  topk_kernel<<<kB, 256, 0, stream>>>(hsp, kp, nzv, nzi, nzc, acc_abs, acc_sq,
                                      acc_nnz);
  dec_kernel<<<kB, 256, 0, stream>>>(wdT, nzv, nzi, nzc, pre_bias, mu, sd,
                                     recon);
  finalize_kernel<<<1, 1, 0, stream>>>(acc_abs, acc_sq, acc_nnz, out3);
}
